// SentimentRNN_80977313399746
// MI455X (gfx1250) — matrix operand order repaired and checked by reading, not compiled, not run on hardware
//
#include <hip/hip_runtime.h>

typedef __attribute__((ext_vector_type(16))) __bf16 v16bf;
typedef __attribute__((ext_vector_type(8)))  float  v8f;

#define HID   512
#define EDIM  512
#define NG4   2048      // 4 gates * 512
#define NTOK  4096      // 32 sentences * 128 tokens
#define NSENT 32
#define SLEN  128
#define REC_BLOCKS 32

static __device__ __forceinline__ unsigned short f2bf(float f) {
    unsigned u = __builtin_bit_cast(unsigned, f);
    unsigned r = u + 0x7FFFu + ((u >> 16) & 1u);   // round-to-nearest-even
    return (unsigned short)(r >> 16);
}

union FragB { v16bf v; unsigned d[8]; };

// ---------------- packing / gather kernels ----------------

__global__ void k_pack_w(const float* __restrict__ Wi, const float* __restrict__ Wf,
                         const float* __restrict__ Wo, const float* __restrict__ Wc,
                         unsigned short* __restrict__ WT) {
    int idx = blockIdx.x * blockDim.x + threadIdx.x;        // over 2048*512
    if (idx >= NG4 * HID) return;
    int n = idx >> 9;                                       // packed output column
    int k = idx & 511;                                      // input (K) dim
    int g = n >> 9, j = n & 511;
    const float* W = (g == 0) ? Wi : (g == 1) ? Wf : (g == 2) ? Wo : Wc;
    WT[(size_t)n * HID + k] = f2bf(W[(size_t)k * HID + j]); // transposed: [n][k]
}

__global__ void k_pack_b(const float* __restrict__ bi, const float* __restrict__ bf,
                         const float* __restrict__ bo, const float* __restrict__ bc,
                         float* __restrict__ bx) {
    int n = blockIdx.x * blockDim.x + threadIdx.x;
    if (n >= NG4) return;
    int g = n >> 9, j = n & 511;
    const float* b = (g == 0) ? bi : (g == 1) ? bf : (g == 2) ? bo : bc;
    bx[n] = b[j];
}

__global__ void k_gather(const int* __restrict__ x, const float* __restrict__ emb,
                         unsigned short* __restrict__ Ebf) {
    int idx = blockIdx.x * blockDim.x + threadIdx.x;        // over 4096*512
    if (idx >= NTOK * EDIM) return;
    int t = idx >> 9, k = idx & 511;
    int tok = x[t];
    Ebf[idx] = f2bf(emb[(size_t)tok * EDIM + k]);
}

__global__ void k_zero_bar(unsigned* bar) { bar[0] = 0u; bar[1] = 0u; }

// ---------------- Xg = Ebf @ WxT^T + b  (WMMA bf16 GEMM) ----------------
// One 16x16 output tile per wave; K = 512 in 16 steps of v_wmma_f32_16x16x32_bf16.

__global__ void __launch_bounds__(256) k_xgemm(const unsigned short* __restrict__ Ebf,
                                               const unsigned short* __restrict__ WxT,
                                               const float* __restrict__ bx,
                                               float* __restrict__ Xg) {
    const int wave = threadIdx.x >> 5;
    const int lane = threadIdx.x & 31;
    const int lin  = blockIdx.x * 8 + wave;   // 0 .. 256*128-1 tile id
    const int mT   = lin >> 7;                // 256 M-tiles
    const int nT   = lin & 127;               // 128 N-tiles
    const int l15  = lane & 15;
    const int hi   = (lane >> 4) & 1;

    const unsigned short* arow = Ebf + (size_t)(mT * 16 + l15) * HID;
    const unsigned short* brow = WxT + (size_t)(nT * 16 + l15) * HID;
    __builtin_prefetch(arow, 0, 0);
    __builtin_prefetch(brow, 0, 0);

    v8f acc = {};
#pragma unroll
    for (int ks = 0; ks < 16; ++ks) {
        FragB a, b;
#pragma unroll
        for (int r = 0; r < 8; ++r) {
            // A (16x32, M x K): K = 16*(r/4) + 8*hi + 2*(r%4) + h
            int ka = ks * 32 + ((r >> 2) << 4) + (hi << 3) + ((r & 3) << 1);
            a.d[r] = *(const unsigned*)(arow + ka);
            // B (32x16, K x N): K = 16*(r/4) + 8*hi + 2*(r%4) + h  (WxT is [n][k], contiguous in k)
            int kb = ks * 32 + ((r >> 2) << 4) + (hi << 3) + ((r & 3) << 1);
            b.d[r] = *(const unsigned*)(brow + kb);
        }
        acc = __builtin_amdgcn_wmma_f32_16x16x32_bf16(false, a.v, false, b.v,
                                                      (short)0, acc, false, false);
    }
    const float bias = bx[nT * 16 + l15];
#pragma unroll
    for (int e = 0; e < 8; ++e) {
        int row = mT * 16 + e + (hi << 3);    // C: vgpr e -> M=e (lanes 0-15), M=8+e (16-31)
        int col = nT * 16 + l15;
        Xg[(size_t)row * NG4 + col] = acc[e] + bias;
    }
}

// ---------------- persistent recurrence kernel ----------------

static __device__ __forceinline__ void gridsync(unsigned* bar) {
    __syncthreads();
    if (threadIdx.x == 0) {
        unsigned* cnt = bar;
        unsigned* gen = bar + 1;
        unsigned my = __hip_atomic_load(gen, __ATOMIC_RELAXED, __HIP_MEMORY_SCOPE_AGENT);
        unsigned arr = __hip_atomic_fetch_add(cnt, 1u, __ATOMIC_ACQ_REL, __HIP_MEMORY_SCOPE_AGENT);
        if (arr == REC_BLOCKS - 1) {
            __hip_atomic_store(cnt, 0u, __ATOMIC_RELAXED, __HIP_MEMORY_SCOPE_AGENT);
            __hip_atomic_fetch_add(gen, 1u, __ATOMIC_RELEASE, __HIP_MEMORY_SCOPE_AGENT);
        } else {
            while (__hip_atomic_load(gen, __ATOMIC_ACQUIRE, __HIP_MEMORY_SCOPE_AGENT) == my)
                __builtin_amdgcn_s_sleep(1);
        }
    }
    __syncthreads();
}

// 32 blocks x 128 threads. Block w owns hidden dims [16w, 16w+16); wave g = gate g.
// Each wave keeps its 512x16 W_h slice resident in VGPRs as 16 WMMA B-fragments.
__global__ void __launch_bounds__(128) k_rec(const unsigned short* __restrict__ WhT,
                                             const float* __restrict__ Xg,
                                             const float* __restrict__ h0,
                                             const float* __restrict__ c0,
                                             unsigned short* __restrict__ hbf,  // 2 x 512 bf16
                                             float* __restrict__ Hs,            // 32 x 512
                                             unsigned* bar) {
    const int w    = blockIdx.x;
    const int lane = threadIdx.x & 31;
    const int g    = threadIdx.x >> 5;
    const int l15  = lane & 15;
    const int hi   = (lane >> 4) & 1;

    FragB bfr[16];
    {
        const unsigned short* brow = WhT + (size_t)(g * HID + w * 16 + l15) * HID;
#pragma unroll
        for (int ks = 0; ks < 16; ++ks)
#pragma unroll
            for (int r = 0; r < 8; ++r)
                bfr[ks].d[r] = *(const unsigned*)(brow + ks * 32 + ((r >> 2) << 4) + (hi << 3) + ((r & 3) << 1));
    }

    __shared__ float C_s[16];
    __shared__ float G_s[4][16];

    if (threadIdx.x < 16) {
        C_s[threadIdx.x] = c0[31 * HID + w * 16 + threadIdx.x];
        hbf[w * 16 + threadIdx.x] = f2bf(h0[31 * HID + w * 16 + threadIdx.x]);
    }
    __threadfence();
    gridsync(bar);

    for (int t = 0; t < NTOK; ++t) {
        const unsigned short* hp = hbf + (t & 1) * HID;
        v8f acc = {};
#pragma unroll
        for (int ks = 0; ks < 16; ++ks) {
            FragB a;
#pragma unroll
            for (int r = 0; r < 8; ++r) {
                int ka = ks * 32 + ((r >> 2) << 4) + (hi << 3) + ((r & 3) << 1);
                a.d[r] = *(const unsigned*)(hp + ka);   // all A rows = h (broadcast)
            }
            acc = __builtin_amdgcn_wmma_f32_16x16x32_bf16(false, a.v, false, bfr[ks].v,
                                                          (short)0, acc, false, false);
        }
        if (lane < 16)   // C row 0: vgpr 0, lanes 0-15 = N 0-15
            G_s[g][lane] = acc[0] + Xg[(size_t)t * NG4 + g * HID + w * 16 + lane];
        __syncthreads();

        if (threadIdx.x < 16) {
            int d = threadIdx.x;
            float ii = 1.f / (1.f + __expf(-G_s[0][d]));
            float ff = 1.f / (1.f + __expf(-G_s[1][d]));
            float oo = 1.f / (1.f + __expf(-G_s[2][d]));
            float ct = tanhf(G_s[3][d]);
            float cn = ff * C_s[d] + ii * ct;
            C_s[d] = cn;
            float hn = oo * tanhf(cn);
            hbf[((t + 1) & 1) * HID + w * 16 + d] = f2bf(hn);
            if ((t & (SLEN - 1)) == SLEN - 1)
                Hs[(size_t)(t >> 7) * HID + w * 16 + d] = hn;
        }
        __threadfence();
        gridsync(bar);
    }
}

// ---------------- output projections (small, f32 exact) ----------------

__global__ void k_proj1(const float* __restrict__ Hs, const float* __restrict__ Whq,
                        const float* __restrict__ bq, float* __restrict__ Y) {
    int idx = blockIdx.x * blockDim.x + threadIdx.x;   // 32*512
    if (idx >= NSENT * HID) return;
    int s = idx >> 9, d = idx & 511;
    float acc = bq[d];
    const float* hr = Hs + (size_t)s * HID;
    for (int k = 0; k < HID; ++k) acc += hr[k] * Whq[(size_t)k * HID + d];
    Y[idx] = acc;
}

__global__ void k_proj2(const float* __restrict__ Y, const float* __restrict__ fcW,
                        const float* __restrict__ fcb, float* __restrict__ out) {
    int idx = blockIdx.x * blockDim.x + threadIdx.x;   // 32*5
    if (idx >= NSENT * 5) return;
    int s = idx / 5, o = idx % 5;
    float acc = fcb[o];
    const float* yr = Y + (size_t)s * HID;
    for (int d = 0; d < HID; ++d) acc += yr[d] * fcW[(size_t)o * HID + d];
    out[idx] = acc;
}

// ---------------- launch ----------------

extern "C" void kernel_launch(void* const* d_in, const int* in_sizes, int n_in,
                              void* d_out, int out_size, void* d_ws, size_t ws_size,
                              hipStream_t stream) {
    const int*   x    = (const int*)  d_in[0];
    const float* h0   = (const float*)d_in[1];
    const float* c0   = (const float*)d_in[2];
    const float* emb  = (const float*)d_in[3];
    const float* W_xi = (const float*)d_in[4];
    const float* W_hi = (const float*)d_in[5];
    const float* b_i  = (const float*)d_in[6];
    const float* W_xf = (const float*)d_in[7];
    const float* W_hf = (const float*)d_in[8];
    const float* b_f  = (const float*)d_in[9];
    const float* W_xo = (const float*)d_in[10];
    const float* W_ho = (const float*)d_in[11];
    const float* b_o  = (const float*)d_in[12];
    const float* W_xc = (const float*)d_in[13];
    const float* W_hc = (const float*)d_in[14];
    const float* b_c  = (const float*)d_in[15];
    const float* W_hq = (const float*)d_in[16];
    const float* b_q  = (const float*)d_in[17];
    const float* fc_W = (const float*)d_in[18];
    const float* fc_b = (const float*)d_in[19];
    float* out = (float*)d_out;

    char* p = (char*)d_ws;
    auto carve = [&](size_t bytes) { void* r = (void*)p; p += (bytes + 255) & ~(size_t)255; return r; };
    unsigned short* Ebf = (unsigned short*)carve((size_t)NTOK * EDIM * 2);
    unsigned short* WxT = (unsigned short*)carve((size_t)NG4 * HID * 2);
    unsigned short* WhT = (unsigned short*)carve((size_t)NG4 * HID * 2);
    float*          bx  = (float*)         carve((size_t)NG4 * 4);
    float*          Xg  = (float*)         carve((size_t)NTOK * NG4 * 4);
    unsigned short* hbf = (unsigned short*)carve((size_t)2 * HID * 2);
    float*          Hs  = (float*)         carve((size_t)NSENT * HID * 4);
    float*          Yb  = (float*)         carve((size_t)NSENT * HID * 4);
    unsigned*       bar = (unsigned*)      carve(256);

    k_zero_bar<<<1, 1, 0, stream>>>(bar);
    k_pack_w<<<(NG4 * HID + 255) / 256, 256, 0, stream>>>(W_xi, W_xf, W_xo, W_xc, WxT);
    k_pack_w<<<(NG4 * HID + 255) / 256, 256, 0, stream>>>(W_hi, W_hf, W_ho, W_hc, WhT);
    k_pack_b<<<(NG4 + 255) / 256, 256, 0, stream>>>(b_i, b_f, b_o, b_c, bx);
    k_gather<<<(NTOK * EDIM + 255) / 256, 256, 0, stream>>>(x, emb, Ebf);
    k_xgemm<<<(256 * 128) / 8, 256, 0, stream>>>(Ebf, WxT, bx, Xg);
    k_rec<<<REC_BLOCKS, 128, 0, stream>>>(WhT, Xg, h0, c0, hbf, Hs, bar);
    k_proj1<<<(NSENT * HID + 255) / 256, 256, 0, stream>>>(Hs, W_hq, b_q, Yb);
    k_proj2<<<1, 256, 0, stream>>>(Yb, fc_W, fc_b, out);
    (void)in_sizes; (void)n_in; (void)out_size; (void)ws_size;
}
